// BlockPermProduct_7954279432747
// MI455X (gfx1250) — compile-verified, matrix operand order checked by reading, not run on hardware
//
#include <hip/hip_runtime.h>
#include <hip/hip_bf16.h>
#include <math.h>

typedef __attribute__((ext_vector_type(16))) _Float16 v16h;
typedef __attribute__((ext_vector_type(8)))  _Float16 v8h;
typedef __attribute__((ext_vector_type(8)))  float    v8f;

#define BATCH 65536
#define SIZE  1024
#define NSTAGE 9

// ---------------------------------------------------------------------------
// Kernel 1: compose the 9 linear stages into a single 1024x1024 operator.
// Block k transforms basis vector e_k through all stages in LDS (fp32),
// then writes column k of the TRANSPOSED f16 matrix Mt[n*1024 + k], so the
// GEMM's B-fragments (column n, contiguous k) are contiguous 16B loads.
// ---------------------------------------------------------------------------
__global__ void build_transform_kernel(const float* __restrict__ logits,
                                       _Float16* __restrict__ Mt) {
    __shared__ float buf0[SIZE];
    __shared__ float buf1[SIZE];
    __shared__ float pr[NSTAGE * 3];

    const int tid = threadIdx.x;          // 256 threads
    const int k   = blockIdx.x;           // basis index 0..1023

    if (tid < NSTAGE * 3) {
        float z = logits[tid];
        pr[tid] = 1.0f / (1.0f + __expf(-z));
    }
    for (int j = tid; j < SIZE; j += 256)
        buf0[j] = (j == k) ? 1.0f : 0.0f;
    __syncthreads();

    // stages i = 8..0, block size n = 4<<i (reference loop order)
    for (int i = NSTAGE - 1; i >= 0; --i) {
        const int   n  = 4 << i;
        const int   h  = n >> 1;
        const float p0 = pr[i * 3 + 0];
        const float p1 = pr[i * 3 + 1];
        const float p2 = pr[i * 3 + 2];

        // pass 1: even/odd separation mix
        for (int j = tid; j < SIZE; j += 256) {
            int blk = j & ~(n - 1);
            int pos = j & (n - 1);
            int src = (pos < h) ? (2 * pos) : (2 * (pos - h) + 1);
            buf1[j] = (1.0f - p0) * buf0[j] + p0 * buf0[blk + src];
        }
        __syncthreads();

        // pass 2: probabilistic half reversals
        for (int j = tid; j < SIZE; j += 256) {
            int   blk = j & ~(n - 1);
            int   pos = j & (n - 1);
            int   partner;
            float w;
            if (pos < h) { partner = h - 1 - pos;     w = p1; }
            else         { partner = 3 * h - 1 - pos; w = p2; }
            buf0[j] = (1.0f - w) * buf1[j] + w * buf1[blk + partner];
        }
        __syncthreads();
    }

    // Row k of operator A lives in buf0; scatter into transposed f16 matrix.
    for (int nidx = tid; nidx < SIZE; nidx += 256)
        Mt[(size_t)nidx * SIZE + k] = (_Float16)buf0[nidx];
}

// ---------------------------------------------------------------------------
// Kernel 2: OUT[65536x1024] = X[65536x1024] * A via v_wmma_f32_16x16x32_f16.
// Workgroup tile 128(M) x 128(N), BK=32, 8 waves (4 in M x 2 in N),
// each wave computes 32x64 = 2x4 WMMA accumulators.
// Global->LDS staging uses CDNA5 GLOBAL_LOAD_ASYNC_TO_LDS_B128 (ASYNCcnt),
// double-buffered: tile k+1 copies in flight while tile k feeds the WMMAs.
// ---------------------------------------------------------------------------
#define BM 128
#define BN 128
#define BK 32
#define NK (SIZE / BK)
#define LDA_F 36   // floats per sA row (32 + 4 pad): 144B, 16B aligned
#define LDB_H 40   // halves per sB row (32 + 8 pad):  80B, 16B aligned

static __device__ __forceinline__ unsigned lds_off(const void* p) {
    // addrspace(3) object through a generic pointer: low 32 bits = LDS offset
    return (unsigned)(unsigned long long)(uintptr_t)p;
}

static __device__ __forceinline__ void async_b128(unsigned lds, const void* gp) {
    asm volatile("global_load_async_to_lds_b128 %0, %1, off"
                 :: "v"(lds), "v"((unsigned long long)(uintptr_t)gp)
                 : "memory");
}

__global__ __launch_bounds__(256)
void wmma_gemm_kernel(const float* __restrict__ X,
                      const _Float16* __restrict__ Mt,
                      float* __restrict__ Out) {
    // double-buffered raw tiles: A stays fp32 (converted at fragment build)
    __shared__ __align__(16) float    sA[2][BM * LDA_F];   // 2 x 18 KB
    __shared__ __align__(16) _Float16 sB[2][BN * LDB_H];   // 2 x 10 KB

    const int tid    = threadIdx.x;
    const int lane   = tid & 31;
    const int waveId = tid >> 5;          // 0..7
    const int wm     = waveId & 3;        // 4 waves along M (32 rows each)
    const int wn     = waveId >> 2;       // 2 waves along N (64 cols each)

    const int m0 = blockIdx.y * BM;       // 512 blocks in M
    const int n0 = blockIdx.x * BN;       // 8 blocks in N

    const int r = lane & 15;              // fragment row/col within 16
    const int h = lane >> 4;              // K-half select per ISA layout

    v8f acc[2][4];
#pragma unroll
    for (int mi = 0; mi < 2; ++mi)
#pragma unroll
        for (int ni = 0; ni < 4; ++ni)
            acc[mi][ni] = (v8f)(0.0f);

    // per-thread async chunk coordinates (16B granules)
    // A tile: 1024 chunks (128 rows x 8), 4 per thread
    // B tile:  512 chunks (128 rows x 4), 2 per thread
    auto issue_tile = [&](int kk, int buf) {
        const int kbase = kk * BK;
#pragma unroll
        for (int t = 0; t < 4; ++t) {
            const int cc  = tid + t * 256;        // 0..1023
            const int row = cc >> 3;
            const int kc  = cc & 7;               // x4 floats
            async_b128(lds_off(&sA[buf][row * LDA_F + kc * 4]),
                       X + (size_t)(m0 + row) * SIZE + kbase + kc * 4);
        }
#pragma unroll
        for (int t = 0; t < 2; ++t) {
            const int cc  = tid + t * 256;        // 0..511
            const int row = cc >> 2;
            const int kc  = cc & 3;               // x8 halves
            async_b128(lds_off(&sB[buf][row * LDB_H + kc * 8]),
                       Mt + (size_t)(n0 + row) * SIZE + kbase + kc * 8);
        }
    };

    issue_tile(0, 0);   // prologue: tile 0 in flight

    for (int kk = 0; kk < NK; ++kk) {
        const int cur = kk & 1;

        if (kk + 1 < NK) {
            issue_tile(kk + 1, cur ^ 1);                       // 6 new in flight
            asm volatile("s_wait_asynccnt 0x6" ::: "memory");  // tile kk landed
        } else {
            asm volatile("s_wait_asynccnt 0x0" ::: "memory");
        }
        __syncthreads();   // all waves' copies for tile kk visible

        // --- A fragments: fp32 in LDS -> f16 v16h (ISA 7.12.2 16x32 layout)
        v16h afrag[2];
#pragma unroll
        for (int mi = 0; mi < 2; ++mi) {
            const float* pa = &sA[cur][(wm * 32 + mi * 16 + r) * LDA_F + h * 8];
            float4 q0 = *(const float4*)(pa + 0);
            float4 q1 = *(const float4*)(pa + 4);
            float4 q2 = *(const float4*)(pa + 16);   // K = 16 + h*8 ...
            float4 q3 = *(const float4*)(pa + 20);
            v16h a;
            a[0]  = (_Float16)q0.x; a[1]  = (_Float16)q0.y;
            a[2]  = (_Float16)q0.z; a[3]  = (_Float16)q0.w;
            a[4]  = (_Float16)q1.x; a[5]  = (_Float16)q1.y;
            a[6]  = (_Float16)q1.z; a[7]  = (_Float16)q1.w;
            a[8]  = (_Float16)q2.x; a[9]  = (_Float16)q2.y;
            a[10] = (_Float16)q2.z; a[11] = (_Float16)q2.w;
            a[12] = (_Float16)q3.x; a[13] = (_Float16)q3.y;
            a[14] = (_Float16)q3.z; a[15] = (_Float16)q3.w;
            afrag[mi] = a;
        }

        // --- B fragments: f16 operator tile, k-contiguous 16B loads
        v16h bfrag[4];
#pragma unroll
        for (int ni = 0; ni < 4; ++ni) {
            const _Float16* pb = &sB[cur][(wn * 64 + ni * 16 + r) * LDB_H + h * 8];
            v8h lo = *(const v8h*)(pb);
            v8h hi = *(const v8h*)(pb + 16);
            bfrag[ni] = __builtin_shufflevector(lo, hi,
                0, 1, 2, 3, 4, 5, 6, 7, 8, 9, 10, 11, 12, 13, 14, 15);
        }

#pragma unroll
        for (int mi = 0; mi < 2; ++mi)
#pragma unroll
            for (int ni = 0; ni < 4; ++ni)
                acc[mi][ni] = __builtin_amdgcn_wmma_f32_16x16x32_f16(
                    false, afrag[mi], false, bfrag[ni],
                    (short)0, acc[mi][ni], false, false);

        __syncthreads();   // tile kk fully consumed; its buffer may be refilled
    }

    // --- epilogue: C/D layout — VGPR v holds row (v + 8*h), lane r = column
#pragma unroll
    for (int mi = 0; mi < 2; ++mi) {
#pragma unroll
        for (int ni = 0; ni < 4; ++ni) {
            const int colg = n0 + wn * 64 + ni * 16 + r;
#pragma unroll
            for (int v = 0; v < 8; ++v) {
                const int rowg = m0 + wm * 32 + mi * 16 + v + 8 * h;
                Out[(size_t)rowg * SIZE + colg] = acc[mi][ni][v];
            }
        }
    }
}

// ---------------------------------------------------------------------------
extern "C" void kernel_launch(void* const* d_in, const int* in_sizes, int n_in,
                              void* d_out, int out_size, void* d_ws, size_t ws_size,
                              hipStream_t stream) {
    const float* X      = (const float*)d_in[0];   // 65536 x 1024 fp32
    const float* logits = (const float*)d_in[1];   // 9 x 3 fp32
    float*       Out    = (float*)d_out;           // 65536 x 1024 fp32
    _Float16*    Mt     = (_Float16*)d_ws;         // 1024 x 1024 f16 (2 MB)

    // 1) Compose the 9 mixing stages into one 1024x1024 f16 operator (transposed).
    build_transform_kernel<<<SIZE, 256, 0, stream>>>(logits, Mt);

    // 2) OUT = X * A via WMMA f16 (f32 accumulate), async-LDS double buffered.
    dim3 grid(SIZE / BN, BATCH / BM);   // (8, 512)
    wmma_gemm_kernel<<<grid, 256, 0, stream>>>(X, Mt, Out);
}
